// SLSTM2_78915729097279
// MI455X (gfx1250) — compile-verified, matrix operand order checked by reading, not run on hardware
//
#include <hip/hip_runtime.h>

// CDNA5 / gfx1250 sequential-LSTM kernel.
// Strategy: batch-tiled recurrence. 64 workgroups x 16 batch columns.
// Recurrent weights (4 gates x 128x128) live as bf16 WMMA A-fragments in
// VGPRs for the whole T=512 loop; h/c state lives in LDS; 16 WMMAs per wave
// per timestep (v_wmma_f32_16x16x32_bf16).

typedef __attribute__((ext_vector_type(16))) __bf16 v16bf;
typedef __attribute__((ext_vector_type(8)))  __bf16 v8bf;
typedef __attribute__((ext_vector_type(8)))  float  v8f;

#define Bsz 1024
#define Tsz 512
#define Isz 2
#define Hsz 128
#define BT  16   // batch columns per workgroup

__device__ __forceinline__ float fsig(float x) {
    return 1.0f / (1.0f + __expf(-x));            // v_exp_f32 + div
}
__device__ __forceinline__ float ftanh(float x) {
    return 2.0f * fsig(2.0f * x) - 1.0f;
}

__global__ __launch_bounds__(256, 1) void slstm_wmma_kernel(
    const float* __restrict__ X,
    const float* __restrict__ wf_x, const float* __restrict__ wf_h,
    const float* __restrict__ wi_x, const float* __restrict__ wi_h,
    const float* __restrict__ wo_x, const float* __restrict__ wo_h,
    const float* __restrict__ wc_x, const float* __restrict__ wc_h,
    const float* __restrict__ bf_,  const float* __restrict__ bi_,
    const float* __restrict__ bo_,  const float* __restrict__ bc_,
    const float* __restrict__ wy,   const float* __restrict__ by,
    float* __restrict__ out)
{
    __shared__ __bf16 h_lds[BT * Hsz];          // [n][k] col-major, bf16 (B-frag source)
    __shared__ float  c_lds[BT * Hsz];          // [n][m]
    __shared__ float  gates_lds[4 * BT * Hsz];  // [gate][n][m]
    __shared__ float  wx_lds[4 * Hsz * Isz];    // [gate][m][i]
    __shared__ float  b_lds[4 * Hsz];
    __shared__ float  wy_lds[Hsz];
    __shared__ float  part_lds[BT * 16];        // [n][mgroup] y-reduction

    const int tid  = threadIdx.x;
    const int lane = tid & 31;
    const int wave = tid >> 5;
    const int gate = wave >> 1;       // 0=f 1=i 2=o 3=c
    const int half = wave & 1;
    const int rowbase = half * 64;
    const int lm   = lane & 15;       // M row (A) / N col (B,C)
    const int lhi  = lane >> 4;       // lane-half selects K sub-range
    const int kbase = lhi * 8;        // K base within a 32-wide K tile
    const int bcol = blockIdx.x * BT;

    const float* whp[4] = { wf_h, wi_h, wo_h, wc_h };
    const float* wxp[4] = { wf_x, wi_x, wo_x, wc_x };
    const float* bp[4]  = { bf_,  bi_,  bo_,  bc_  };

    // ---- one-time init of LDS-resident small operands + state ----
    #pragma unroll
    for (int g = 0; g < 4; ++g) {
        wx_lds[g * (Hsz * Isz) + tid] = wxp[g][tid];          // 256 each
        if (tid < Hsz) b_lds[g * Hsz + tid] = bp[g][tid];
    }
    if (tid < Hsz) wy_lds[tid] = wy[tid];
    for (int j = tid; j < BT * Hsz; j += 256) {
        h_lds[j] = (__bf16)0.0f;                               // h0 = 0
        c_lds[j] = 0.0f;                                       // c0 = 0
    }
    const float by0 = by[0];

    // ---- recurrent weights -> bf16 A-fragments, VGPR-resident for all T ----
    // A 16x32 bf16 fragment: lane = M row; lane-half picks K {0..7,16..23} vs
    // {8..15,24..31} (ISA 7.12.2 16-bit A layout).
    v16bf afrag[4][4];                 // [Mtile][Ktile]
    const float* whg = whp[gate];
    #pragma unroll
    for (int i = 0; i < 4; ++i) {
        #pragma unroll
        for (int kk = 0; kk < 4; ++kk) {
            const int m = rowbase + i * 16 + lm;
            const float* src = whg + m * Hsz + kk * 32 + kbase;
            v16bf a;
            #pragma unroll
            for (int j = 0; j < 8; ++j) {
                a[j]     = (__bf16)src[j];        // K = kbase + j
                a[8 + j] = (__bf16)src[16 + j];   // K = kbase + 16 + j
            }
            afrag[i][kk] = a;
        }
    }

    __syncthreads();

    const int en  = tid & 15;          // elementwise: batch column
    const int emg = tid >> 4;          // elementwise: m-group (8 rows each)
    const int em0 = emg * 8;

    for (int t = 0; t < Tsz; ++t) {
        // ---- B fragments of h (16B-aligned ds loads, same K-split as A) ----
        v16bf bfrag[4];
        #pragma unroll
        for (int kk = 0; kk < 4; ++kk) {
            const __bf16* hb = &h_lds[lm * Hsz + kk * 32 + kbase];
            v8bf lo = *(const v8bf*)hb;            // K = kbase + 0..7
            v8bf hi = *(const v8bf*)(hb + 16);     // K = kbase + 16..23
            v16bf b;
            #pragma unroll
            for (int j = 0; j < 8; ++j) { b[j] = lo[j]; b[8 + j] = hi[j]; }
            bfrag[kk] = b;
        }

        // x_t for this lane's batch column (8B-aligned, I=2 contiguous)
        const float2 xv = *(const float2*)&X[(size_t)(bcol + lm) * (Tsz * Isz) + t * Isz];

        // ---- 16 WMMAs: gate-half pre-activations, then + W_x x_t + bias ----
        #pragma unroll
        for (int i = 0; i < 4; ++i) {
            v8f acc = {};
            #pragma unroll
            for (int kk = 0; kk < 4; ++kk) {
                acc = __builtin_amdgcn_wmma_f32_16x16x32_bf16(
                    false, afrag[i][kk], false, bfrag[kk],
                    (short)0, acc, false, false);
            }
            const int mb = rowbase + i * 16 + lhi * 8;   // C layout: VGPR r -> M
            #pragma unroll
            for (int r = 0; r < 8; ++r) {
                const int m = mb + r;
                float v = acc[r]
                        + wx_lds[gate * (Hsz * Isz) + 2 * m]     * xv.x
                        + wx_lds[gate * (Hsz * Isz) + 2 * m + 1] * xv.y
                        + b_lds[gate * Hsz + m];
                gates_lds[gate * (BT * Hsz) + lm * Hsz + m] = v;
            }
        }

        __syncthreads();   // gates visible to all waves

        // ---- elementwise cell update: 8 (m,n) elements per thread ----
        float part = 0.0f;
        #pragma unroll
        for (int j = 0; j < 8; ++j) {
            const int m   = em0 + j;
            const int idx = en * Hsz + m;
            const float fg = fsig (gates_lds[0 * (BT * Hsz) + idx]);
            const float ig = fsig (gates_lds[1 * (BT * Hsz) + idx]);
            const float og = fsig (gates_lds[2 * (BT * Hsz) + idx]);
            const float gg = ftanh(gates_lds[3 * (BT * Hsz) + idx]);
            const float c  = fg * c_lds[idx] + ig * gg;
            c_lds[idx] = c;
            const float h  = og * ftanh(c);
            h_lds[idx] = (__bf16)h;                 // next step's B operand
            part += wy_lds[m] * h;
        }
        part_lds[en * 16 + emg] = part;

        __syncthreads();   // h/c/partials stable before next GEMM + y output

        if (tid < BT) {
            float y = by0;
            #pragma unroll
            for (int j = 0; j < 16; ++j) y += part_lds[tid * 16 + j];
            out[(size_t)t * Bsz + bcol + tid] = y;   // ys[t, 0, b]
        }
    }
}

extern "C" void kernel_launch(void* const* d_in, const int* in_sizes, int n_in,
                              void* d_out, int out_size, void* d_ws, size_t ws_size,
                              hipStream_t stream) {
    const float* X    = (const float*)d_in[0];
    const float* wf_x = (const float*)d_in[1];
    const float* wf_h = (const float*)d_in[2];
    const float* wi_x = (const float*)d_in[3];
    const float* wi_h = (const float*)d_in[4];
    const float* wo_x = (const float*)d_in[5];
    const float* wo_h = (const float*)d_in[6];
    const float* wc_x = (const float*)d_in[7];
    const float* wc_h = (const float*)d_in[8];
    const float* bf_  = (const float*)d_in[9];
    const float* bi_  = (const float*)d_in[10];
    const float* bo_  = (const float*)d_in[11];
    const float* bc_  = (const float*)d_in[12];
    const float* wy   = (const float*)d_in[13];
    const float* by   = (const float*)d_in[14];
    float* out = (float*)d_out;

    slstm_wmma_kernel<<<dim3(Bsz / BT), dim3(256), 0, stream>>>(
        X, wf_x, wf_h, wi_x, wi_h, wo_x, wo_h, wc_x, wc_h,
        bf_, bi_, bo_, bc_, wy, by, out);
}
